// PointerNetwork_19172734009902
// MI455X (gfx1250) — matrix operand order repaired and checked by reading, not compiled, not run on hardware
//
#include <hip/hip_runtime.h>
#include <hip/hip_bf16.h>

// ---------------------------------------------------------------------------
// Pointer network decoder for gfx1250 (MI455X), wave32 + WMMA bf16.
//
//   1) prep_weights : fp32 -> bf16 weight conversion + bias(=bih+bhh)
//   2) build_xcat   : gather word_emb[tokens] and concat with enc, in bf16
//   3) gemm_bf16    : WMMA GEMMs, software-pipelined K loop (ping-pong regs)
//        Xproj = Xcat @ Wih^T + (bih+bhh)   [2048 x 2048]
//        b1    = enc  @ W1^T                [2048 x 256]
//        b2    = emb  @ W2^T                [2048 x 256]
//        b3    = emb  @ W3^T                [2048 x 256]
//   4) decode_kernel: 1 workgroup / batch element; serial loop. Recurrence
//        g = Xproj[p] + Whh@h via WMMA; h held in 128 VGPRs (broadcast into
//        all 16 B columns, loaded from LDS once per step).
// ---------------------------------------------------------------------------

#define B_  8
#define L_  256
#define E_  256
#define H_  256
#define W_  256
#define EH_ 512
#define IN_ 768
#define G4_ 2048          // 4*EH
#define BL_ 2048          // B*L
#define NEG_INF (-1e30f)

typedef __attribute__((ext_vector_type(8)))  __bf16 v8bf;
typedef __attribute__((ext_vector_type(16))) __bf16 v16bf;
typedef __attribute__((ext_vector_type(8)))  float  v8f;

#define BFCAT(lo, hi) __builtin_shufflevector(lo, hi, 0,1,2,3,4,5,6,7,8,9,10,11,12,13,14,15)

__device__ __forceinline__ unsigned short f2bf(float f) {
  union { float f; unsigned u; } v; v.f = f;
  unsigned u = v.u;
  unsigned r = u + 0x7FFFu + ((u >> 16) & 1u);   // round-to-nearest-even
  return (unsigned short)(r >> 16);
}
__device__ __forceinline__ float sigm(float x) { return 1.f / (1.f + __expf(-x)); }

// ---------------------------------------------------------------------------
// 1) weight conversion
// ---------------------------------------------------------------------------
__global__ void prep_weights(const float* __restrict__ Wih, const float* __restrict__ Whh,
                             const float* __restrict__ W1,  const float* __restrict__ W2,
                             const float* __restrict__ W3,
                             const float* __restrict__ bih, const float* __restrict__ bhh,
                             unsigned short* __restrict__ WihB, unsigned short* __restrict__ WhhB,
                             unsigned short* __restrict__ W1B,  unsigned short* __restrict__ W2B,
                             unsigned short* __restrict__ W3B,  float* __restrict__ bias) {
  int i = blockIdx.x * blockDim.x + threadIdx.x;
  int stride = gridDim.x * blockDim.x;
  for (int idx = i; idx < G4_ * IN_; idx += stride) {
    WihB[idx] = f2bf(Wih[idx]);
    if (idx < G4_ * EH_) WhhB[idx] = f2bf(Whh[idx]);
    if (idx < W_ * EH_)  W1B[idx]  = f2bf(W1[idx]);
    if (idx < W_ * E_) { W2B[idx]  = f2bf(W2[idx]); W3B[idx] = f2bf(W3[idx]); }
    if (idx < G4_)       bias[idx] = bih[idx] + bhh[idx];
  }
}

// ---------------------------------------------------------------------------
// 2) Xcat[r, 0:256]=emb(word), [256:768]=enc   (bf16)
// ---------------------------------------------------------------------------
__global__ void build_xcat(const int* __restrict__ tokens, const float* __restrict__ word_emb,
                           const float* __restrict__ enc, unsigned short* __restrict__ Xcat) {
  int r = blockIdx.x, t = threadIdx.x;            // r in [0, B*L)
  int tok = tokens[r];
  Xcat[(size_t)r * IN_ + t]           = f2bf(word_emb[(size_t)tok * E_ + t]);
  Xcat[(size_t)r * IN_ + E_ + t]      = f2bf(enc[(size_t)r * EH_ + t]);
  Xcat[(size_t)r * IN_ + E_ + H_ + t] = f2bf(enc[(size_t)r * EH_ + H_ + t]);
}

// ---------------------------------------------------------------------------
// 3) GEMM: C[M,N] = A[M,K](bf16, lda) @ Bw[N,K]^T (bf16) + bias[N]
//    one wave computes a 16x64 tile; K loop ping-pong software pipelined
//    (requires K % 64 == 0; K is 768 / 512 / 256 here).
// ---------------------------------------------------------------------------
__global__ void __launch_bounds__(256)
gemm_bf16_wmma(const unsigned short* __restrict__ A, int lda,
               const unsigned short* __restrict__ Bw,
               const float* __restrict__ bias,
               float* __restrict__ C, int M, int N, int K) {
  const int lane = threadIdx.x & 31;
  const int gw   = (blockIdx.x * blockDim.x + threadIdx.x) >> 5;
  const int nsupers = N >> 6;
  const int mt = gw / nsupers;
  const int ns = gw - mt * nsupers;
  if (mt >= (M >> 4)) return;
  const int m0 = mt << 4, n0 = ns << 6;
  const int h8  = (lane >> 4) * 8;      // A per-lane K sub-chunk select
  const int h16 = (lane >> 4) * 16;     // unused; ldB also selects K with h8
  const unsigned short* arow = A + (size_t)(m0 + (lane & 15)) * lda;
  const unsigned short* bcol = Bw + (size_t)(n0 + (lane & 15)) * K;

  auto ldA = [&](int k0) -> v16bf {
    v8bf lo = *(const v8bf*)(arow + k0 + h8);
    v8bf hi = *(const v8bf*)(arow + k0 + 16 + h8);
    return BFCAT(lo, hi);
  };
  auto ldB = [&](int k0, int n) -> v16bf {
    const unsigned short* q = bcol + (size_t)(16 * n) * K + k0 + h8;
    v8bf lo = *(const v8bf*)q;
    v8bf hi = *(const v8bf*)(q + 16);
    return BFCAT(lo, hi);
  };

  v8f acc[4] = { {}, {}, {}, {} };
  v16bf A0, A1, B0[4], B1[4];

  A0 = ldA(0);
#pragma unroll
  for (int n = 0; n < 4; ++n) B0[n] = ldB(0, n);

  for (int k0 = 0; k0 < K; k0 += 64) {
    // preload chunk k0+32 while computing chunk k0
    __builtin_prefetch(arow + k0 + 128, 0, 1);    // global_prefetch_b8
    A1 = ldA(k0 + 32);
#pragma unroll
    for (int n = 0; n < 4; ++n) B1[n] = ldB(k0 + 32, n);
#pragma unroll
    for (int n = 0; n < 4; ++n)
      acc[n] = __builtin_amdgcn_wmma_f32_16x16x32_bf16(
          false, A0, false, B0[n], (short)0, acc[n], false, false);
    // preload chunk k0+64 while computing chunk k0+32
    if (k0 + 64 < K) {
      A0 = ldA(k0 + 64);
#pragma unroll
      for (int n = 0; n < 4; ++n) B0[n] = ldB(k0 + 64, n);
    }
#pragma unroll
    for (int n = 0; n < 4; ++n)
      acc[n] = __builtin_amdgcn_wmma_f32_16x16x32_bf16(
          false, A1, false, B1[n], (short)0, acc[n], false, false);
  }

  const int rbase = m0 + h8;
#pragma unroll
  for (int n = 0; n < 4; ++n) {
    const int col = n0 + (lane & 15) + 16 * n;
    const float badd = bias ? bias[col] : 0.f;
#pragma unroll
    for (int j = 0; j < 8; ++j)
      C[(size_t)(rbase + j) * N + col] = acc[n][j] + badd;
  }
}

// ---------------------------------------------------------------------------
// 4) serial decode: 8 blocks (1/batch), 256 threads = 8 waves.
//    Only 2 waves/SIMD -> huge VGPR budget; keep broadcast h in 128 VGPRs.
// ---------------------------------------------------------------------------

// g[0:2048] = Whh(bf16)[2048,512] @ h(bf16 LDS, broadcast into all 16 B cols)
__device__ __forceinline__ void matvec_whh(const unsigned short* __restrict__ Wbf,
                                           const unsigned short* hb, float* g,
                                           int lane, int wave) {
  const int h8  = (lane >> 4) * 8;
  const int h16 = (lane >> 4) * 16;

  // hoist the h-broadcast B operands: 16 chunks x 8 VGPRs, loaded once/step
  v16bf Bh[16];
#pragma unroll
  for (int kk = 0; kk < 16; ++kk) {
    v8bf lo = *(const v8bf*)(hb + kk * 32 + h8);
    v8bf hi = *(const v8bf*)(hb + kk * 32 + 16 + h8);
    Bh[kk] = BFCAT(lo, hi);
  }

#pragma unroll 1
  for (int t = 0; t < 16; ++t) {
    const int m0 = wave * 256 + t * 16;
    const unsigned short* arow = Wbf + (size_t)(m0 + (lane & 15)) * EH_;
    v8f acc = {};
#pragma unroll
    for (int kk = 0; kk < 16; ++kk) {
      v8bf alo = *(const v8bf*)(arow + kk * 32 + h8);
      v8bf ahi = *(const v8bf*)(arow + kk * 32 + 16 + h8);
      acc = __builtin_amdgcn_wmma_f32_16x16x32_bf16(
          false, BFCAT(alo, ahi), false, Bh[kk], (short)0, acc, false, false);
    }
    if ((lane & 15) == 0) {                         // all 16 columns identical
      const int base = m0 + (lane >> 4) * 8;
#pragma unroll
      for (int j = 0; j < 8; ++j) g[base + j] = acc[j];
    }
  }
}

__device__ __forceinline__ void lstm_gates(const float* sg, const float* __restrict__ xp,
                                           float* sh, float* scell, unsigned short* shb,
                                           int tid) {
  for (int e = tid; e < EH_; e += 256) {
    float gi = sg[e]           + xp[e];
    float gf = sg[EH_ + e]     + xp[EH_ + e];
    float gg = sg[2 * EH_ + e] + xp[2 * EH_ + e];
    float go = sg[3 * EH_ + e] + xp[3 * EH_ + e];
    float c2 = sigm(gf) * scell[e] + sigm(gi) * tanhf(gg);
    float hv = sigm(go) * tanhf(c2);
    scell[e] = c2; sh[e] = hv; shb[e] = f2bf(hv);
  }
}

__global__ void __launch_bounds__(256)
decode_kernel(const float* __restrict__ enc, const float* __restrict__ W4,
              const float* __restrict__ vt1, const float* __restrict__ bias,
              const unsigned short* __restrict__ WhhB,
              const float* __restrict__ Xproj,
              const float* __restrict__ b1, const float* __restrict__ b2,
              const float* __restrict__ b3, float* __restrict__ out) {
  const int bId = blockIdx.x;
  const int tid = threadIdx.x;
  const int lane = tid & 31, wave = tid >> 5;

  __shared__ float sh[EH_], scell[EH_], sg[G4_], sw4h[W_], ssc[L_];
  __shared__ __align__(16) unsigned short shb[EH_];
  __shared__ int ssi[L_];

  const float* encB = enc + (size_t)bId * L_ * EH_;

  // h0 = concat(enc[-1,:H], enc[0,H:]); c0 = 0
  for (int e = tid; e < EH_; e += 256) {
    float v = (e < H_) ? encB[(size_t)(L_ - 1) * EH_ + e] : encB[e];
    sh[e] = v; shb[e] = f2bf(v); scell[e] = 0.f;
  }
  out[(size_t)bId * L_ + tid] = (tid == 0) ? 1.f : 0.f;
  __syncthreads();

  // initial LSTM step with zero input: g = Whh@h0 + (bih+bhh)
  matvec_whh(WhhB, shb, sg, lane, wave);
  __syncthreads();
  lstm_gates(sg, bias, sh, scell, shb, tid);
  __syncthreads();

  int bond = 0, ptr = 0, last = 0;
  // 1 initial attend + <= 2L body steps (matches reference scan budget)
  for (int step = 0; step <= 2 * L_; ++step) {
    bool do_attend = (step == 0) || (ptr >= bond);
    if (!do_attend) {
      // ---- LSTM token step: g = Xproj[b, ptr] + Whh@h ----
      matvec_whh(WhhB, shb, sg, lane, wave);
      __syncthreads();
      lstm_gates(sg, Xproj + ((size_t)bId * L_ + ptr) * G4_, sh, scell, shb, tid);
      __syncthreads();
      ++ptr;
    } else {
      if (step > 0 && tid == 0) out[(size_t)bId * L_ + bond] = 1.f;
      // ---- attention: argmax_l>=bond  tanh(b1[l]+b2[l]+b3[bond]+W4h) . vt1 ----
      {  // w4h = W4 @ h
        const float* wr = W4 + (size_t)tid * EH_;
        float a = 0.f;
        for (int k = 0; k < EH_; ++k) a += wr[k] * sh[k];
        sw4h[tid] = a;
      }
      __syncthreads();
      {
        const float* r1 = b1 + ((size_t)bId * L_ + tid) * W_;
        const float* r2 = b2 + ((size_t)bId * L_ + tid) * W_;
        const float* r3 = b3 + ((size_t)bId * L_ + bond) * W_;
        float a = 0.f;
        for (int w = 0; w < W_; ++w)
          a += vt1[w] * tanhf(r1[w] + r2[w] + r3[w] + sw4h[w]);
        ssc[tid] = (tid >= bond) ? a : NEG_INF;   // mask; softmax shift is argmax-invariant
        ssi[tid] = tid;
      }
      __syncthreads();
      for (int s = 128; s > 0; s >>= 1) {          // first-max reduction
        if (tid < s) {
          float x0 = ssc[tid], x1 = ssc[tid + s];
          int   i1 = ssi[tid + s];
          if (x1 > x0 || (x1 == x0 && i1 < ssi[tid])) { ssc[tid] = x1; ssi[tid] = i1; }
        }
        __syncthreads();
      }
      int nb = ssi[0] + 1;
      __syncthreads();
      last = bond; bond = nb; (void)last;
      if (bond >= L_) break;                       // uniform
    }
  }
}

// ---------------------------------------------------------------------------
extern "C" void kernel_launch(void* const* d_in, const int* in_sizes, int n_in,
                              void* d_out, int out_size, void* d_ws, size_t ws_size,
                              hipStream_t stream) {
  (void)in_sizes; (void)n_in; (void)out_size; (void)ws_size;
  const int*   tokens   = (const int*)d_in[0];
  const float* enc      = (const float*)d_in[1];
  const float* word_emb = (const float*)d_in[2];
  // d_in[3] chunk_emb and d_in[9] vt2: add a uniform shift to all logits ->
  // argmax-invariant, unused.
  const float* W1  = (const float*)d_in[4];
  const float* W2  = (const float*)d_in[5];
  const float* W3  = (const float*)d_in[6];
  const float* W4  = (const float*)d_in[7];
  const float* vt1 = (const float*)d_in[8];
  const float* Wih = (const float*)d_in[10];
  const float* Whh = (const float*)d_in[11];
  const float* bih = (const float*)d_in[12];
  const float* bhh = (const float*)d_in[13];
  float* out = (float*)d_out;

  char* base = (char*)d_ws;
  size_t off = 0;
  auto take = [&](size_t bytes) -> char* {
    char* q = base + off;
    off += bytes; off = (off + 255) & ~(size_t)255;
    return q;
  };
  unsigned short* Xcat = (unsigned short*)take((size_t)BL_ * IN_ * 2);
  unsigned short* WihB = (unsigned short*)take((size_t)G4_ * IN_ * 2);
  unsigned short* WhhB = (unsigned short*)take((size_t)G4_ * EH_ * 2);
  unsigned short* W1B  = (unsigned short*)take((size_t)W_ * EH_ * 2);
  unsigned short* W2B  = (unsigned short*)take((size_t)W_ * E_ * 2);
  unsigned short* W3B  = (unsigned short*)take((size_t)W_ * E_ * 2);
  float* bias  = (float*)take((size_t)G4_ * 4);
  float* Xproj = (float*)take((size_t)BL_ * G4_ * 4);
  float* b1    = (float*)take((size_t)BL_ * W_ * 4);
  float* b2    = (float*)take((size_t)BL_ * W_ * 4);
  float* b3    = (float*)take((size_t)BL_ * W_ * 4);

  prep_weights<<<1024, 256, 0, stream>>>(Wih, Whh, W1, W2, W3, bih, bhh,
                                         WihB, WhhB, W1B, W2B, W3B, bias);
  build_xcat<<<BL_, 256, 0, stream>>>(tokens, word_emb, enc, Xcat);

  {   // Xproj = Xcat @ Wih^T + (bih+bhh)
    int waves = (BL_ / 16) * (G4_ / 64);
    gemm_bf16_wmma<<<waves / 8, 256, 0, stream>>>(Xcat, IN_, WihB, bias, Xproj, BL_, G4_, IN_);
  }
  {   // b1 = enc @ W1^T (enc slice of Xcat, lda = IN); b2/b3 from emb slice
    int waves = (BL_ / 16) * (W_ / 64);
    gemm_bf16_wmma<<<waves / 8, 256, 0, stream>>>(Xcat + E_, IN_, W1B, nullptr, b1, BL_, W_, EH_);
    gemm_bf16_wmma<<<waves / 8, 256, 0, stream>>>(Xcat, IN_, W2B, nullptr, b2, BL_, W_, E_);
    gemm_bf16_wmma<<<waves / 8, 256, 0, stream>>>(Xcat, IN_, W3B, nullptr, b3, BL_, W_, E_);
  }
  decode_kernel<<<B_, 256, 0, stream>>>(enc, W4, vt1, bias, WhhB, Xproj, b1, b2, b3, out);
}
